// VNGNN_39024072851537
// MI455X (gfx1250) — matrix operand order repaired and checked by reading, not compiled, not run on hardware
//
#include <hip/hip_runtime.h>
#include <hip/hip_bf16.h>

#define NNODES  100000
#define NEDGES  1600000
#define KDIM    128
#define EPS_BN  1e-5f
#define NSLOPE  0.2f

#if defined(__HIP_DEVICE_COMPILE__) && \
    __has_builtin(__builtin_amdgcn_global_load_async_to_lds_b128) && \
    __has_builtin(__builtin_amdgcn_s_wait_asynccnt)
#define HAVE_ASYNC_LDS 1
#else
#define HAVE_ASYNC_LDS 0
#endif

typedef __attribute__((ext_vector_type(16))) __bf16 v16bf;
typedef __attribute__((ext_vector_type(8)))  __bf16 v8bf;
typedef __attribute__((ext_vector_type(8)))  float  v8f;
typedef __attribute__((ext_vector_type(4)))  int    v4i;

#define AS_GLOBAL __attribute__((address_space(1)))
#define AS_LDS    __attribute__((address_space(3)))

__device__ __forceinline__ float lrelu(float v) { return v > 0.f ? v : NSLOPE * v; }

// float atomic max via signed/unsigned int ordering trick (valid incl. -inf init)
__device__ __forceinline__ void atomicMaxF(float* addr, float val) {
  if (val >= 0.f) atomicMax((int*)addr, __float_as_int(val));
  else            atomicMin((unsigned int*)addr, __float_as_uint(val));
}

// ---------------- elementwise / conversion ----------------
__global__ void k_f32_to_bf16(const float* __restrict__ x, __bf16* __restrict__ y, int n) {
  int i = blockIdx.x * blockDim.x + threadIdx.x;
  if (i < n) y[i] = (__bf16)x[i];
}

// wt[n][k] = W[k][n] as bf16; pad columns [ncols, npad) with zero
__global__ void k_build_wt(const float* __restrict__ w, __bf16* __restrict__ wt,
                           int ncols, int npad) {
  int i = blockIdx.x * blockDim.x + threadIdx.x;
  if (i >= npad * KDIM) return;
  int n = i >> 7, k = i & 127;
  wt[(size_t)n * KDIM + k] = (n < ncols) ? (__bf16)w[(size_t)k * ncols + n] : (__bf16)0.f;
}

__global__ void k_zero_f32(float* __restrict__ p, int n) {
  int i = blockIdx.x * blockDim.x + threadIdx.x;
  if (i < n) p[i] = 0.f;
}

__global__ void k_init_md(float* __restrict__ m, float* __restrict__ den, int n) {
  int i = blockIdx.x * blockDim.x + threadIdx.x;
  if (i < n) { m[i] = -__builtin_inff(); den[i] = 0.f; }
}

__global__ void k_fix_m(float* __restrict__ m, int n) {
  int i = blockIdx.x * blockDim.x + threadIdx.x;
  if (i < n) { float v = m[i]; m[i] = (v < -1e37f) ? 0.f : v; }
}

__global__ void k_init_out(float* __restrict__ out, const float* __restrict__ b, int n) {
  int i = blockIdx.x * blockDim.x + threadIdx.x;
  if (i < n) out[i] = b[i % 40];
}

// ---------------- WMMA GEMM: out[N x (NT*16), ld=LDC] = Xbf[N x 128] * Wt^T ----------------
// Block = 8 waves; Wt tile (NT*16 x 128, bf16) is staged once into LDS (async-to-LDS when
// available), rows padded to 136 bf16 so ds_load_b128 fragments are bank-conflict-free.
// One wave computes a 16 x (NT*16) strip: A fragment loaded once per K-step from global,
// B fragments from LDS, NT independent accumulators -> back-to-back v_wmma.
// Requires nrows % 16 == 0 (true: 100000), so no per-row store guards.
template <int LDC, int NT>
__global__ void k_gemm_wmma(const __bf16* __restrict__ xb, const __bf16* __restrict__ wt,
                            float* __restrict__ out, int nrows) {
  constexpr int KP = KDIM + 8;              // padded LDS row stride (bf16 units)
  __shared__ __bf16 Bs[NT * 16 * KP];
  const int tid = threadIdx.x;

  // ---- stage Wt tile into LDS (all waves participate, before any early exit) ----
#if HAVE_ASYNC_LDS
  for (int c = tid; c < NT * 16 * 16; c += 256) {       // 16B chunks: row-major Wt
    int row = c >> 4, off = c & 15;
    __builtin_amdgcn_global_load_async_to_lds_b128(
        (AS_GLOBAL v4i*)(wt + (size_t)c * 8),
        (AS_LDS v4i*)(Bs + row * KP + off * 8),
        0, 0);
  }
  __builtin_amdgcn_s_wait_asynccnt(0);
#else
  for (int c = tid; c < NT * 16 * 16; c += 256) {
    int row = c >> 4, off = c & 15;
    *(uint4*)(Bs + row * KP + off * 8) = ((const uint4*)wt)[c];
  }
#endif
  __syncthreads();

  int wave  = tid >> 5;
  int lane  = tid & 31;
  int tileM = blockIdx.x * 8 + wave;
  int row0  = tileM << 4;
  if (row0 >= nrows) return;
  int lhalf = lane >> 4;
  int l16   = lane & 15;

  // A frag (16-bit 16x32): lane L (half=L>>4): VGPR0-3 = K klo..klo+7, VGPR4-7 = K klo+16..+23
  const __bf16* ap = xb + (size_t)(row0 + l16) * KDIM + lhalf * 8;

  v8f acc[NT];
#pragma unroll
  for (int nt = 0; nt < NT; ++nt) { v8f z = {}; acc[nt] = z; }

#pragma unroll
  for (int kt = 0; kt < 4; ++kt) {
    union { v16bf v; v8bf h[2]; } a;
    a.h[0] = *(const v8bf*)(ap + kt * 32);        // K klo   .. klo+7
    a.h[1] = *(const v8bf*)(ap + kt * 32 + 16);   // K klo+16.. klo+23
#pragma unroll
    for (int nt = 0; nt < NT; ++nt) {
      // B frag: lane holds column n = nt*16 + l16; same K order as A (klo..klo+7, klo+16..klo+23)
      const __bf16* bsp = Bs + (nt * 16 + l16) * KP + lhalf * 8;
      union { v16bf v; v8bf h[2]; } b;
      b.h[0] = *(const v8bf*)(bsp + kt * 32);      // K klo   .. klo+7
      b.h[1] = *(const v8bf*)(bsp + kt * 32 + 16); // K klo+16.. klo+23
      acc[nt] = __builtin_amdgcn_wmma_f32_16x16x32_bf16(false, a.v, false, b.v,
                                                        (short)0, acc[nt], false, false);
    }
  }

  // C layout: VGPR j -> row j (lanes 0-15) / row j+8 (lanes 16-31), col = lane&15
  float* op = out + (size_t)(row0 + lhalf * 8) * LDC + l16;
#pragma unroll
  for (int nt = 0; nt < NT; ++nt)
#pragma unroll
    for (int j = 0; j < 8; ++j)
      op[j * LDC + nt * 16] = acc[nt][j];
}

// ---------------- attention logits ----------------
__global__ void k_logits(const float* __restrict__ h, const float* __restrict__ asrc,
                         const float* __restrict__ adst, float* __restrict__ als,
                         float* __restrict__ ald, int nH, int H, int D, int ld) {
  int t = blockIdx.x * blockDim.x + threadIdx.x;
  if (t >= nH) return;
  int node = t / H, hd = t - node * H;
  const float* hp = h + (size_t)node * ld + hd * D;
  const float* as = asrc + hd * D;
  const float* ad = adst + hd * D;
  float s = 0.f, d2 = 0.f;
  for (int i = 0; i < D; ++i) { float v = hp[i]; s += v * as[i]; d2 += v * ad[i]; }
  als[t] = s; ald[t] = d2;
}

// ---------------- edge passes ----------------
__global__ void k_edge_max(const long long* __restrict__ ei, const float* __restrict__ als,
                           const float* __restrict__ ald, float* __restrict__ m, int H) {
  int e = blockIdx.x * blockDim.x + threadIdx.x;
  if (e >= NEDGES) return;
  int s = (int)ei[e], d = (int)ei[NEDGES + e];
  for (int hd = 0; hd < H; ++hd) {
    float v = lrelu(als[s * H + hd] + ald[d * H + hd]);
    atomicMaxF(&m[d * H + hd], v);
  }
}

__global__ void k_edge_sum(const long long* __restrict__ ei, const float* __restrict__ als,
                           const float* __restrict__ ald, const float* __restrict__ m,
                           float* __restrict__ den, int H) {
  int e = blockIdx.x * blockDim.x + threadIdx.x;
  if (e >= NEDGES) return;
  int s = (int)ei[e], d = (int)ei[NEDGES + e];
  for (int hd = 0; hd < H; ++hd) {
    float v = lrelu(als[s * H + hd] + ald[d * H + hd]);
    atomicAdd(&den[d * H + hd], __expf(v - m[d * H + hd]));
  }
}

// one wave per edge; 128 channels (H=2, D=64); float4 gather + 4 atomics/lane
__global__ void k_edge_aggr128(const long long* __restrict__ ei, const float* __restrict__ h,
                               const float* __restrict__ als, const float* __restrict__ ald,
                               const float* __restrict__ m, const float* __restrict__ den,
                               float* __restrict__ out) {
  int e = blockIdx.x * 8 + (threadIdx.x >> 5);
  if (e >= NEDGES) return;
  int lane = threadIdx.x & 31;
  int s = (int)ei[e], d = (int)ei[NEDGES + e];
  int hd = lane >> 4;                                 // lanes 0-15: head 0, 16-31: head 1
  float v = lrelu(als[s * 2 + hd] + ald[d * 2 + hd]);
  float alpha = __expf(v - m[d * 2 + hd]) / (den[d * 2 + hd] + 1e-16f);
  const float4 hv = *(const float4*)(h + (size_t)s * 128 + lane * 4);
  float* op = out + (size_t)d * 128 + lane * 4;
  atomicAdd(op + 0, alpha * hv.x);
  atomicAdd(op + 1, alpha * hv.y);
  atomicAdd(op + 2, alpha * hv.z);
  atomicAdd(op + 3, alpha * hv.w);
}

// one wave per edge; 40 channels (H=1), h stride 48, out stride 40
__global__ void k_edge_aggr40(const long long* __restrict__ ei, const float* __restrict__ h,
                              const float* __restrict__ als, const float* __restrict__ ald,
                              const float* __restrict__ m, const float* __restrict__ den,
                              float* __restrict__ out) {
  int e = blockIdx.x * 8 + (threadIdx.x >> 5);
  if (e >= NEDGES) return;
  int lane = threadIdx.x & 31;
  int s = (int)ei[e], d = (int)ei[NEDGES + e];
  float v = lrelu(als[s] + ald[d]);
  float alpha = __expf(v - m[d]) / (den[d] + 1e-16f);
  for (int c = lane; c < 40; c += 32)
    atomicAdd(&out[(size_t)d * 40 + c], alpha * h[(size_t)s * 48 + c]);
}

// ---------------- batch norm ----------------
__global__ void k_bn_stats(const float* __restrict__ x, float* __restrict__ stats, int nrows) {
  __shared__ float ss[128], sq[128];
  int ch = threadIdx.x & 127, half = threadIdx.x >> 7;
  int r0 = blockIdx.x * 512;
  int rend = r0 + 512; if (rend > nrows) rend = nrows;
  float s = 0.f, q = 0.f;
  for (int r = r0 + half; r < rend; r += 2) {
    float v = x[(size_t)r * 128 + ch]; s += v; q += v * v;
  }
  if (half) { ss[ch] = s; sq[ch] = q; }
  __syncthreads();
  if (!half) {
    atomicAdd(&stats[ch],       s + ss[ch]);
    atomicAdd(&stats[128 + ch], q + sq[ch]);
  }
}

__global__ void k_bn_coef(const float* __restrict__ stats, const float* __restrict__ g,
                          const float* __restrict__ beta, float* __restrict__ coef, int nrows) {
  int ch = threadIdx.x;
  if (ch >= 128) return;
  float mu  = stats[ch] / (float)nrows;
  float var = stats[128 + ch] / (float)nrows - mu * mu;
  float sc  = g[ch] * rsqrtf(var + EPS_BN);
  coef[ch] = sc;
  coef[128 + ch] = beta[ch] - mu * sc;
}

__global__ void k_bn_apply(const float* __restrict__ x, const float* __restrict__ coef,
                           __bf16* __restrict__ yb, int n) {
  int i = blockIdx.x * blockDim.x + threadIdx.x;
  if (i >= n) return;
  int ch = i & 127;
  float v = x[i] * coef[ch] + coef[128 + ch];
  v = v > 0.f ? v : 0.f;
  yb[i] = (__bf16)v;
}

// ---------------- orchestration ----------------
static inline dim3 G(int n) { return dim3((unsigned)((n + 255) / 256)); }

extern "C" void kernel_launch(void* const* d_in, const int* in_sizes, int n_in,
                              void* d_out, int out_size, void* d_ws, size_t ws_size,
                              hipStream_t stream) {
  const float*     x     = (const float*)d_in[0];
  const long long* ei    = (const long long*)d_in[1];
  const float*     w0    = (const float*)d_in[2];
  const float*     asrc0 = (const float*)d_in[3];
  const float*     adst0 = (const float*)d_in[4];
  const float*     g0    = (const float*)d_in[6];
  const float*     beta0 = (const float*)d_in[7];
  const float*     w1    = (const float*)d_in[8];
  const float*     asrc1 = (const float*)d_in[9];
  const float*     adst1 = (const float*)d_in[10];
  const float*     g1    = (const float*)d_in[12];
  const float*     beta1 = (const float*)d_in[13];
  const float*     w2    = (const float*)d_in[14];
  const float*     asrc2 = (const float*)d_in[15];
  const float*     adst2 = (const float*)d_in[16];
  const float*     b2    = (const float*)d_in[17];
  float*           out   = (float*)d_out;

  char* ws = (char*)d_ws;
  __bf16* XB   = (__bf16*)(ws);                                  // N*128 bf16
  __bf16* WT   = (__bf16*)(ws + 25600000);                       // 128*128 bf16 (padded)
  float*  HB   = (float*)(ws + 25632768);                        // N*128 f32 (layer2: N*48)
  float*  AGG  = (float*)(ws + 76832768);                        // N*128 f32
  float*  ALS  = (float*)(ws + 128032768);                       // N*2 f32
  float*  ALD  = (float*)(ws + 128832768);
  float*  MB   = (float*)(ws + 129632768);
  float*  DEN  = (float*)(ws + 130432768);
  float*  ST   = (float*)(ws + 131232768);                       // 512 f32 (stats + coef)

  const int mblocks = (NNODES + 127) / 128;                      // 782 (8 M-tiles/block)
  const dim3 eblk((NEDGES + 7) / 8);                             // wave-per-edge kernels

  // ---- layer 0 input -> bf16 ----
  k_f32_to_bf16<<<G(NNODES * 128), 256, 0, stream>>>(x, XB, NNODES * 128);

  for (int layer = 0; layer < 2; ++layer) {
    const float* w    = layer ? w1 : w0;
    const float* as_  = layer ? asrc1 : asrc0;
    const float* ad_  = layer ? adst1 : adst0;
    const float* g_   = layer ? g1 : g0;
    const float* be_  = layer ? beta1 : beta0;

    k_build_wt<<<G(128 * KDIM), 256, 0, stream>>>(w, WT, 128, 128);
    k_gemm_wmma<128, 8><<<dim3(mblocks), 256, 0, stream>>>(XB, WT, HB, NNODES);
    k_logits<<<G(NNODES * 2), 256, 0, stream>>>(HB, as_, ad_, ALS, ALD, NNODES * 2, 2, 64, 128);
    k_init_md<<<G(NNODES * 2), 256, 0, stream>>>(MB, DEN, NNODES * 2);
    k_edge_max<<<G(NEDGES), 256, 0, stream>>>(ei, ALS, ALD, MB, 2);
    k_fix_m<<<G(NNODES * 2), 256, 0, stream>>>(MB, NNODES * 2);
    k_edge_sum<<<G(NEDGES), 256, 0, stream>>>(ei, ALS, ALD, MB, DEN, 2);
    k_zero_f32<<<G(NNODES * 128), 256, 0, stream>>>(AGG, NNODES * 128);
    k_edge_aggr128<<<eblk, 256, 0, stream>>>(ei, HB, ALS, ALD, MB, DEN, AGG);
    k_zero_f32<<<1, 256, 0, stream>>>(ST, 256);
    k_bn_stats<<<(NNODES + 511) / 512, 256, 0, stream>>>(AGG, ST, NNODES);
    k_bn_coef<<<1, 128, 0, stream>>>(ST, g_, be_, ST + 256, NNODES);
    k_bn_apply<<<G(NNODES * 128), 256, 0, stream>>>(AGG, ST + 256, XB, NNODES * 128);
  }

  // ---- layer 2: GATConv(128 -> 40, heads=1), no BN/ReLU ----
  k_build_wt<<<G(48 * KDIM), 256, 0, stream>>>(w2, WT, 40, 48);
  k_gemm_wmma<48, 3><<<dim3(mblocks), 256, 0, stream>>>(XB, WT, HB, NNODES);
  k_logits<<<G(NNODES), 256, 0, stream>>>(HB, asrc2, adst2, ALS, ALD, NNODES, 1, 40, 48);
  k_init_md<<<G(NNODES), 256, 0, stream>>>(MB, DEN, NNODES);
  k_edge_max<<<G(NEDGES), 256, 0, stream>>>(ei, ALS, ALD, MB, 1);
  k_fix_m<<<G(NNODES), 256, 0, stream>>>(MB, NNODES);
  k_edge_sum<<<G(NEDGES), 256, 0, stream>>>(ei, ALS, ALD, MB, DEN, 1);
  k_init_out<<<G(NNODES * 40), 256, 0, stream>>>(out, b2, NNODES * 40);
  k_edge_aggr40<<<eblk, 256, 0, stream>>>(ei, HB, ALS, ALD, MB, DEN, out);
}
